// RNN_MultiRegional_STRALM_30356828848603
// MI455X (gfx1250) — matrix operand order repaired and checked by reading, not compiled, not run on hardware
//
#include <hip/hip_runtime.h>

// ---------------------------------------------------------------------------
// RNN_MultiRegional_STRALM on MI455X (gfx1250)
// Persistent 16-workgroup kernel; TC-scaled bf16 weights resident in 320KB LDS;
// fp32 state tiles pinned in VGPRs; bf16 state ping-pongs through L2;
// v_wmma_f32_16x16x32_bf16 for the recurrent matmul; inhib/inp streams staged
// one step ahead via global_load_async_to_lds_b128 (ASYNCcnt) + L2 prefetch;
// per-step sync = atomic sense barrier + CDNA5 cluster barrier.
// ---------------------------------------------------------------------------

#define Bsz     32
#define Tsteps  2000
#define HH      512
#define NH      1024          // 2H
#define TCv     0.01f
#define NWG     16
#define NCOL    64            // output columns per workgroup
#define THREADS 256
#define WSTRIDE 1032          // padded LDS row stride (elements) -> 4-bank shift/row
#define ISTRIDE 68            // padded row stride for f32 staging buffers

typedef __attribute__((ext_vector_type(16))) __bf16 v16bf;
typedef __attribute__((ext_vector_type(8)))  __bf16 v8bf;
typedef __attribute__((ext_vector_type(8)))  float  v8f;

// d_out layout (float elements), reference return order:
// (hn_last[1,32,1024], rnn_out[32,2000,1024], x_last[1,32,1024], x_out[32,2000,1024])
#define OUT_HN    0L
#define OUT_RNN   (32L * 1024L)
#define OUT_XLAST (OUT_RNN + 32L * 2000L * 1024L)
#define OUT_XOUT  (OUT_XLAST + 32L * 1024L)

// workspace layout (bytes): hb0 | hb1 | barrier{count,sense}
#define WS_HB0 0
#define WS_HB1 65536
#define WS_BAR 131072

// dynamic LDS layout (bytes)
#define SM_W     0                           // 64 x 1032 bf16          = 132096
#define SM_STG   132096                      // 32 x 68  f32            = 8704
#define SM_WI    (SM_STG + 8704)             // 4 x 64   f32            = 1024
#define SM_MSK   (SM_WI + 1024)              // 64       f32            = 256
#define SM_IB    (SM_MSK + 256)              // 2 x 32 x 68 f32 (inhib) = 17408
#define SM_INP   (SM_IB + 17408)             // 2 x 32 x 4  f32 (inp)   = 1024
#define SMEM_BYTES (SM_INP + 1024)           // 160512

typedef __attribute__((address_space(3))) const char* lds_cp;

__device__ __forceinline__ unsigned lds_off(const void* p) {
    return (unsigned)(unsigned long long)(lds_cp)p;
}

// CDNA5 async copy global -> LDS (tracked by ASYNCcnt), 16B per lane.
__device__ __forceinline__ void async_ld_b128(const void* lds, const void* gaddr) {
    unsigned off = lds_off(lds);
    asm volatile("global_load_async_to_lds_b128 %0, %1, off"
                 :: "v"(off), "v"(gaddr) : "memory");
}

__device__ __forceinline__ void wait_async0() {
    asm volatile("s_wait_asynccnt 0x0" ::: "memory");
}

__device__ __forceinline__ void grid_barrier(unsigned* bar, unsigned target) {
    __threadfence();                       // make this thread's global writes visible (agent scope)
    __syncthreads();
    if (threadIdx.x == 0) {
        unsigned prev = __hip_atomic_fetch_add(bar, 1u, __ATOMIC_ACQ_REL,
                                               __HIP_MEMORY_SCOPE_AGENT);
        if (prev == NWG - 1) {
            __hip_atomic_store(bar, 0u, __ATOMIC_RELAXED, __HIP_MEMORY_SCOPE_AGENT);
            __hip_atomic_fetch_add(bar + 1, 1u, __ATOMIC_RELEASE,
                                   __HIP_MEMORY_SCOPE_AGENT);
        } else {
            while (__hip_atomic_load(bar + 1, __ATOMIC_ACQUIRE,
                                     __HIP_MEMORY_SCOPE_AGENT) < target) {
                __builtin_amdgcn_s_sleep(1);
            }
        }
    }
    // CDNA5 cluster barrier (s_barrier_signal -3 / s_barrier_wait -3).
    // NOP when dispatch is not cluster-configured; atomic barrier is authoritative.
    __builtin_amdgcn_s_cluster_barrier();
    __syncthreads();
}

// Issue next step's inhib (8KB/WG) + inp (512B) into LDS double buffer,
// and prefetch the inhib stream 4 steps ahead into L2.
__device__ __forceinline__ void issue_step_loads(char* smem, const float* __restrict__ inhib,
                                                 const float* __restrict__ inp,
                                                 int n0, int tid, int t) {
    const int buf = t & 1;
    float* ib = (float*)(smem + SM_IB) + buf * (Bsz * ISTRIDE);
    float* ip = (float*)(smem + SM_INP) + buf * (Bsz * 4);
    const int r  = tid >> 3;               // batch row 0..31
    const int c0 = (tid & 7) * 8;          // 8-float chunk
    const float* g = inhib + ((long)r * Tsteps + t) * NH + n0 + c0;
    async_ld_b128(ib + r * ISTRIDE + c0,     g);
    async_ld_b128(ib + r * ISTRIDE + c0 + 4, g + 4);
    if (t + 4 < Tsteps)
        __builtin_prefetch(g + 4L * NH, 0, 1);   // HBM -> L2 lookahead
    if (tid < Bsz)
        async_ld_b128(ip + tid * 4, inp + ((long)tid * Tsteps + t) * 4);
}

__global__ void __launch_bounds__(THREADS, 1)
stralm_rnn(const float* __restrict__ inp, const float* __restrict__ hn,
           const float* __restrict__ inhib,
           const float* __restrict__ s2s_w, const float* __restrict__ a2a_w,
           const float* __restrict__ a2s_w, const float* __restrict__ s2a_w,
           const float* __restrict__ inp_w, const float* __restrict__ s2s_f,
           const float* __restrict__ s2s_m, const float* __restrict__ a2s_m,
           const float* __restrict__ a2a_s, const float* __restrict__ d1_m,
           float* __restrict__ out,
           __bf16* __restrict__ hb0, __bf16* __restrict__ hb1,
           unsigned* __restrict__ bar) {
    extern __shared__ char smem[];
    __bf16* Wl  = (__bf16*)(smem + SM_W);    // [64][WSTRIDE] TC-scaled bf16 W slice
    float*  stg = (float*)(smem + SM_STG);   // [32][68] f32 staging
    float*  wiL = (float*)(smem + SM_WI);    // [4][64] inp_weight slice
    float*  mskL= (float*)(smem + SM_MSK);   // [64] str_d1_mask slice

    const int tid  = threadIdx.x;
    const int wg   = blockIdx.x;
    const int n0   = wg * NCOL;
    const int lane = tid & 31;
    const int wave = tid >> 5;
    const int mt   = wave & 1;               // M tile (batch 0-15 / 16-31)
    const int nt   = wave >> 1;              // N tile (0..3) within 64-col slice
    const int m0   = mt * 16;
    const int half = (lane < 16) ? 0 : 1;    // lane half per WMMA layout
    const int lr   = lane & 15;

    // ---- build TC * W_rec[n0..n0+63][:] into LDS as bf16 (one time) ----
    for (int e = tid; e < NCOL * NH; e += THREADS) {
        int r  = e >> 10;            // local row 0..63
        int kc = e & (NH - 1);       // column 0..1023
        int n  = n0 + r;
        float w;
        if (n < HH) {                // striatum rows: [str2str | alm2str]
            if (kc < HH) {
                float v = s2s_w[n * HH + kc];
                v = fminf(fmaxf(v, 1e-15f), 1.0f);
                w = -(s2s_m[n * HH + kc] * v + s2s_f[n * HH + kc]);
            } else {
                int c = kc - HH;
                float v = a2s_w[n * HH + c];
                v = fminf(fmaxf(v, 1e-15f), 1.0f);
                w = a2s_m[n * HH + c] * v;
            }
        } else {                     // ALM rows: [str2alm | alm2alm*sign]
            int rr = n - HH;
            if (kc < HH) {
                float v = s2a_w[rr * HH + kc];
                w = fminf(fmaxf(v, 1e-15f), 1.0f);
            } else {
                int c = kc - HH;
                float v = a2a_w[rr * HH + c];
                v = fminf(fmaxf(v, 1e-15f), 1.0f);
                w = v * a2a_s[c];
            }
        }
        Wl[r * WSTRIDE + kc] = (__bf16)(TCv * w);
    }
    for (int e = tid; e < 4 * NCOL; e += THREADS)
        wiL[e] = inp_w[(e >> 6) * NH + n0 + (e & 63)];
    if (tid < NCOL) mskL[tid] = d1_m[n0 + tid];

    // ---- seed global bf16 state buffer (this WG's slice) ----
    for (int e = tid; e < Bsz * (NCOL / 8); e += THREADS) {
        int r  = e >> 3;
        int c0 = (e & 7) * 8;
        const float* src = hn + r * NH + n0 + c0;
        v8bf v;
        #pragma unroll
        for (int i = 0; i < 8; ++i) v[i] = (__bf16)src[i];
        *(v8bf*)(hb0 + r * NH + n0 + c0) = v;
    }

    // ---- this wave's persistent fp32 state tile (D layout) ----
    const int myN = n0 + nt * 16 + lr;       // global output column
    const int cl  = nt * 16 + lr;            // local column in slice
    v8f hprev;
    #pragma unroll
    for (int j = 0; j < 8; ++j) {
        int m = m0 + j + half * 8;
        hprev[j] = hn[m * NH + myN];
    }

    // kick off async staging of step-0 inhib/inp streams into LDS
    issue_step_loads(smem, inhib, inp, n0, tid, 0);

    unsigned tgt = 1;
    grid_barrier(bar, tgt++);                // LDS W built + hb0 seeded everywhere

    const float w0 = wiL[0 * 64 + cl], w1 = wiL[1 * 64 + cl];
    const float w2 = wiL[2 * 64 + cl], w3 = wiL[3 * 64 + cl];
    const float msk = mskL[cl];

    for (int t = 0; t < Tsteps; ++t) {
        const __bf16* hb_cur = (t & 1) ? hb1 : hb0;
        __bf16*       hb_nxt = (t & 1) ? hb0 : hb1;

        // ---- acc = TC * (h @ W_sliceᵀ) via v_wmma_f32_16x16x32_bf16 ----
        v8f acc = {0.f, 0.f, 0.f, 0.f, 0.f, 0.f, 0.f, 0.f};
        const __bf16* Arow = hb_cur + (m0 + lr) * NH + half * 8;         // A: row m, K split 0/8
        const __bf16* Brow = Wl + (nt * 16 + lr) * WSTRIDE + half * 8;   // B: col n, K split 0/8
        #pragma unroll 4
        for (int k = 0; k < NH; k += 32) {
            v8bf alo = *(const v8bf*)(Arow + k);
            v8bf ahi = *(const v8bf*)(Arow + k + 16);
            v8bf blo = *(const v8bf*)(Brow + k);
            v8bf bhi = *(const v8bf*)(Brow + k + 16);
            v16bf a = __builtin_shufflevector(alo, ahi, 0,1,2,3,4,5,6,7,8,9,10,11,12,13,14,15);
            v16bf b = __builtin_shufflevector(blo, bhi, 0,1,2,3,4,5,6,7,8,9,10,11,12,13,14,15);
            acc = __builtin_amdgcn_wmma_f32_16x16x32_bf16(
                      false, a, false, b, (short)0, acc, false, false);
        }

        // ---- async-staged operands for this step are ready after wait+sync ----
        wait_async0();            // this wave's outstanding async loads (issued last step)
        __syncthreads();          // all waves' async data visible in LDS

        const float* ib = (const float*)(smem + SM_IB) + (t & 1) * (Bsz * ISTRIDE);
        const float* ip = (const float*)(smem + SM_INP) + (t & 1) * (Bsz * 4);

        // ---- epilogue: drive + leaky ReLU update, fp32 state in registers ----
        #pragma unroll
        for (int j = 0; j < 8; ++j) {
            int m = m0 + j + half * 8;
            const float4 f = *(const float4*)(ip + m * 4);   // LDS broadcast read
            float d = (f.x * w0 + f.y * w1 + f.z * w2 + f.w * w3) * msk
                    + ib[m * ISTRIDE + cl];
            float hv = fmaxf(0.0f, (1.0f - TCv) * hprev[j] + acc[j] + TCv * d);
            hprev[j] = hv;
            stg[m * ISTRIDE + cl] = hv;
        }

        // stage next step's streams while writeback + barrier + next K-loop run
        if (t + 1 < Tsteps)
            issue_step_loads(smem, inhib, inp, n0, tid, t + 1);

        __syncthreads();

        // ---- coalesced writeback: rnn_out (fp32) + next bf16 state slice ----
        {
            int r  = tid >> 3;                 // batch row 0..31
            int c0 = (tid & 7) * 8;            // 8-float chunk
            const float* s = stg + r * ISTRIDE + c0;
            float4 lo = *(const float4*)(s);
            float4 hi = *(const float4*)(s + 4);
            long ro = OUT_RNN + ((long)r * Tsteps + t) * NH + n0 + c0;
            *(float4*)(out + ro)     = lo;
            *(float4*)(out + ro + 4) = hi;
            v8bf vb;
            vb[0] = (__bf16)lo.x; vb[1] = (__bf16)lo.y;
            vb[2] = (__bf16)lo.z; vb[3] = (__bf16)lo.w;
            vb[4] = (__bf16)hi.x; vb[5] = (__bf16)hi.y;
            vb[6] = (__bf16)hi.z; vb[7] = (__bf16)hi.w;
            *(v8bf*)(hb_nxt + r * NH + n0 + c0) = vb;
            if (t == Tsteps - 1) {
                long ho = OUT_HN + (long)r * NH + n0 + c0;
                *(float4*)(out + ho)     = lo;
                *(float4*)(out + ho + 4) = hi;
            }
        }

        grid_barrier(bar, tgt++);
    }
}

// x is never updated by the reference: x_out/x_last are broadcasts of x[0].
__global__ void xfill_kernel(const float* __restrict__ x, float* __restrict__ out) {
    const float4* x4 = (const float4*)x;
    float4* xo = (float4*)(out + OUT_XOUT);
    const long total = (long)Bsz * Tsteps * (NH / 4);
    const long stride = (long)gridDim.x * blockDim.x;
    for (long q = (long)blockIdx.x * blockDim.x + threadIdx.x; q < total; q += stride) {
        long bt = q >> 8;                  // NH/4 == 256 float4 per (b,t) row
        int  n4 = (int)(q & 255);
        int  b  = (int)(bt / Tsteps);
        xo[q] = x4[b * 256 + n4];
    }
    long gid = (long)blockIdx.x * blockDim.x + threadIdx.x;
    if (gid < (long)Bsz * (NH / 4))
        ((float4*)(out + OUT_XLAST))[gid] = x4[gid];
}

__global__ void init_bar_kernel(unsigned* bar) { bar[0] = 0u; bar[1] = 0u; }

extern "C" void kernel_launch(void* const* d_in, const int* in_sizes, int n_in,
                              void* d_out, int out_size, void* d_ws, size_t ws_size,
                              hipStream_t stream) {
    const float* inp   = (const float*)d_in[0];
    const float* hn    = (const float*)d_in[1];
    const float* x     = (const float*)d_in[2];
    const float* inhib = (const float*)d_in[3];
    const float* s2s_w = (const float*)d_in[4];
    const float* a2a_w = (const float*)d_in[5];
    const float* a2s_w = (const float*)d_in[6];
    const float* s2a_w = (const float*)d_in[7];
    const float* inp_w = (const float*)d_in[8];
    const float* s2s_f = (const float*)d_in[9];
    const float* s2s_m = (const float*)d_in[10];
    const float* a2s_m = (const float*)d_in[11];
    const float* a2a_s = (const float*)d_in[12];
    const float* d1_m  = (const float*)d_in[13];
    float* out = (float*)d_out;

    char* ws = (char*)d_ws;
    __bf16*   hb0 = (__bf16*)(ws + WS_HB0);
    __bf16*   hb1 = (__bf16*)(ws + WS_HB1);
    unsigned* bar = (unsigned*)(ws + WS_BAR);

    // Allow >64KB dynamic LDS (CDNA5 WGP has 320KB). Idempotent; errors ignored.
    (void)hipFuncSetAttribute((const void*)stralm_rnn,
                              hipFuncAttributeMaxDynamicSharedMemorySize,
                              SMEM_BYTES);

    hipLaunchKernelGGL(init_bar_kernel, dim3(1), dim3(1), 0, stream, bar);
    hipLaunchKernelGGL(xfill_kernel, dim3(2048), dim3(256), 0, stream, x, out);
    hipLaunchKernelGGL(stralm_rnn, dim3(NWG), dim3(THREADS), SMEM_BYTES, stream,
                       inp, hn, inhib, s2s_w, a2a_w, a2s_w, s2a_w, inp_w,
                       s2s_f, s2s_m, a2s_m, a2a_s, d1_m,
                       out, hb0, hb1, bar);
}
